// BEVFormerHead_37658273252096
// MI455X (gfx1250) — matrix operand order repaired and checked by reading, not compiled, not run on hardware
//
#include <hip/hip_runtime.h>
#include <hip/hip_bf16.h>

typedef __attribute__((ext_vector_type(16))) _Float16 v16h;
typedef __attribute__((ext_vector_type(8)))  float    v8f;
typedef __attribute__((ext_vector_type(4)))  float    f4v;
typedef __attribute__((ext_vector_type(4)))  unsigned u4v;

#define LVL 6
#define BB  32
#define QQ  900
#define DD  256
#define NCLS 10

// LDS layout (bytes)
#define XOFF 0            // 32 x 260 f16 = 16640  (cls-branch activations)
#define WOFF 16640        // 256 x 40 f16 = 20480  (transposed weight k-chunk)
#define YOFF 37120        // 32 x 260 f16 = 16640  (reg-branch activations)
#define TOFF 53760        // scratch: LN stats (256B) / head tmp (1280B)
#define LDSZ 55040
// f32 LN staging buffer (32x256 f32 = 32768B) overlays X + first part of W

__device__ __forceinline__ void gemm32x256(
    const _Float16* __restrict__ Xl,   // [32][260] f16
    const float*    __restrict__ Wg,   // [256][256] f32 (level offset applied)
    _Float16*       __restrict__ Wl,   // [256][40] f16 (transposed chunk)
    v8f acc[4], int tid)
{
    const int lane = tid & 31;
    const int w    = tid >> 5;
    const int rt   = w & 1;            // row tile 0..1
    const int ct0  = (w >> 1) << 2;    // first of 4 col tiles
    const int half = lane >> 4;
    const int ln   = lane & 15;

    for (int kc = 0; kc < 8; ++kc) {
        __syncthreads();               // previous Wl consumers done
        // stage transposed weight chunk: Wl[n][k_local] = Wg[kc*32+k][n]
        const float* ws = Wg + (kc * 32) * DD + tid;
        #pragma unroll
        for (int i = 0; i < 32; ++i)
            Wl[tid * 40 + i] = (_Float16)ws[i * DD];
        if (kc < 7) __builtin_prefetch(Wg + (kc + 1) * 32 * DD + tid, 0, 3);
        __syncthreads();

        // A fragment (wave32 16x32 f16): lane<16 -> K 0-15; lane>=16 -> K 16-31 (matches B)
        union { v16h v; unsigned u[8]; } A;
        const _Float16* xr = Xl + (rt * 16 + ln) * 260 + kc * 32;
        #pragma unroll
        for (int i = 0; i < 8; ++i) {
            const int k0 = 2 * i + half * 16;
            A.u[i] = *(const unsigned*)(xr + k0);
        }
        #pragma unroll
        for (int j = 0; j < 4; ++j) {
            // B fragment: lane ln -> N, lanes 16-31 hold K 16-31; contiguous in Wl row
            union { v16h v; u4v q[2]; } Bf;
            const _Float16* wn = Wl + (ct0 + j) * 16 * 40 + ln * 40 + half * 16;
            Bf.q[0] = *(const u4v*)(wn);
            Bf.q[1] = *(const u4v*)(wn + 8);
            acc[j] = __builtin_amdgcn_wmma_f32_16x16x32_f16(
                false, A.v, false, Bf.v, (short)0, acc[j], false, false);
        }
    }
}

__device__ __forceinline__ void store_act_relu(
    const v8f acc[4], _Float16* __restrict__ dst,
    const float* __restrict__ bias, int rt, int ct0, int half, int ln)
{
    #pragma unroll
    for (int j = 0; j < 4; ++j) {
        #pragma unroll
        for (int v = 0; v < 8; ++v) {
            const int row = rt * 16 + v + half * 8;
            const int col = (ct0 + j) * 16 + ln;
            float val = acc[j][v] + bias[col];
            dst[row * 260 + col] = (_Float16)fmaxf(val, 0.f);
        }
    }
}

__device__ __forceinline__ void store_bias_f32(
    const v8f acc[4], float* __restrict__ S,
    const float* __restrict__ bias, int rt, int ct0, int half, int ln)
{
    #pragma unroll
    for (int j = 0; j < 4; ++j) {
        #pragma unroll
        for (int v = 0; v < 8; ++v) {
            const int row = rt * 16 + v + half * 8;
            const int col = (ct0 + j) * 16 + ln;
            S[row * DD + col] = acc[j][v] + bias[col];
        }
    }
}

__device__ __forceinline__ void layernorm_relu(
    const float* __restrict__ S, _Float16* __restrict__ X,
    const float* __restrict__ g, const float* __restrict__ bt,
    float* __restrict__ stats, int tid)
{
    __syncthreads();                       // S fully written
    const int r  = tid >> 3;               // row 0..31, 8 threads/row
    const int c0 = (tid & 7) * 32;
    float vals[32];
    float s = 0.f, sq = 0.f;
    #pragma unroll
    for (int i = 0; i < 32; ++i) {
        const float x = S[r * DD + c0 + i];
        vals[i] = x; s += x; sq += x * x;
    }
    #pragma unroll
    for (int off = 1; off < 8; off <<= 1) {
        s  += __shfl_xor(s,  off, 32);
        sq += __shfl_xor(sq, off, 32);
    }
    if ((tid & 7) == 0) {
        const float mean = s * (1.f / 256.f);
        const float var  = sq * (1.f / 256.f) - mean * mean;
        stats[r * 2 + 0] = mean;
        stats[r * 2 + 1] = rsqrtf(var + 1e-5f);
    }
    __syncthreads();                       // stats ready; all S reads done
    const float mean = stats[r * 2 + 0];
    const float rstd = stats[r * 2 + 1];
    #pragma unroll
    for (int i = 0; i < 32; ++i) {
        const int col = c0 + i;
        const float xn = (vals[i] - mean) * rstd * g[col] + bt[col];
        X[r * 260 + col] = (_Float16)fmaxf(xn, 0.f);   // overlays S (values cached)
    }
}

__device__ __forceinline__ float inv_sigmoid(float x) {
    x = fminf(fmaxf(x, 0.f), 1.f);
    const float x1 = fmaxf(x, 1e-5f);
    const float x2 = fmaxf(1.f - x, 1e-5f);
    return logf(x1 / x2);
}

__global__ __launch_bounds__(256) void bevhead_kernel(
    const float* __restrict__ hs,
    const float* __restrict__ init_ref,
    const float* __restrict__ inter_ref,
    const float* __restrict__ cw1, const float* __restrict__ cb1,
    const float* __restrict__ g1,  const float* __restrict__ b1,
    const float* __restrict__ cw2, const float* __restrict__ cb2,
    const float* __restrict__ g2,  const float* __restrict__ b2,
    const float* __restrict__ cw3, const float* __restrict__ cb3,
    const float* __restrict__ rw1, const float* __restrict__ rb1,
    const float* __restrict__ rw2, const float* __restrict__ rb2,
    const float* __restrict__ rw3, const float* __restrict__ rb3,
    float* __restrict__ out)
{
    __shared__ __align__(16) char lds[LDSZ];
    _Float16* X  = (_Float16*)(lds + XOFF);
    _Float16* Wl = (_Float16*)(lds + WOFF);
    _Float16* Y  = (_Float16*)(lds + YOFF);
    float*    S  = (float*)(lds);                  // f32 LN stage (overlays X+W)
    float*    st = (float*)(lds + TOFF);           // LN stats / head tmp

    const int tid = threadIdx.x;
    const int bid = blockIdx.x;
    const int l   = bid / (QQ * BB / 32);          // bid / 900
    const int m0  = (bid % (QQ * BB / 32)) * 32;

    const int lane = tid & 31, w = tid >> 5;
    const int rt = w & 1, ct0 = (w >> 1) << 2;
    const int half = lane >> 4, ln = lane & 15;

    // ---- load X0 tile (hs is [L,Q,B,D]; logical row m = b*Q + q) ----
    {
        const int r  = tid >> 3;
        const int c0 = (tid & 7) * 32;
        const int gm = m0 + r;
        const int b  = gm / QQ, q = gm % QQ;
        const float* src = hs + (((size_t)l * QQ + q) * BB + b) * DD + c0;
        _Float16* dst = X + r * 260 + c0;
        #pragma unroll
        for (int i = 0; i < 32; i += 4) {
            const f4v f = *(const f4v*)(src + i);
            dst[i + 0] = (_Float16)f.x; dst[i + 1] = (_Float16)f.y;
            dst[i + 2] = (_Float16)f.z; dst[i + 3] = (_Float16)f.w;
        }
    }

    const v8f vzero = {0.f,0.f,0.f,0.f,0.f,0.f,0.f,0.f};
    v8f acc[4];

    // ===== reg layer 1: Y1 = relu(X0 @ rw1 + rb1) =====
    #pragma unroll
    for (int j = 0; j < 4; ++j) acc[j] = vzero;
    gemm32x256(X, rw1 + (size_t)l * DD * DD, Wl, acc, tid);
    __syncthreads();
    store_act_relu(acc, Y, rb1 + l * DD, rt, ct0, half, ln);

    // ===== cls layer 1: X1 = relu(LN(X0 @ cw1 + cb1)) =====
    #pragma unroll
    for (int j = 0; j < 4; ++j) acc[j] = vzero;
    gemm32x256(X, cw1 + (size_t)l * DD * DD, Wl, acc, tid);
    __syncthreads();                               // all reads of X/Wl done
    store_bias_f32(acc, S, cb1 + l * DD, rt, ct0, half, ln);
    layernorm_relu(S, X, g1 + l * DD, b1 + l * DD, st, tid);

    // ===== reg layer 2: Y2 = relu(Y1 @ rw2 + rb2) =====
    #pragma unroll
    for (int j = 0; j < 4; ++j) acc[j] = vzero;
    gemm32x256(Y, rw2 + (size_t)l * DD * DD, Wl, acc, tid);
    __syncthreads();                               // all reads of Y done
    store_act_relu(acc, Y, rb2 + l * DD, rt, ct0, half, ln);

    // ===== cls layer 2: X2 = relu(LN(X1 @ cw2 + cb2)) =====
    #pragma unroll
    for (int j = 0; j < 4; ++j) acc[j] = vzero;
    gemm32x256(X, cw2 + (size_t)l * DD * DD, Wl, acc, tid);
    __syncthreads();
    store_bias_f32(acc, S, cb2 + l * DD, rt, ct0, half, ln);
    layernorm_relu(S, X, g2 + l * DD, b2 + l * DD, st, tid);

    // ===== heads (N=10), scalar dots from LDS =====
    __syncthreads();
    {   // stage cw3 / rw3 (f32) into the free W region
        float* cw3s = (float*)(lds + WOFF);
        float* rw3s = cw3s + DD * NCLS;
        #pragma unroll
        for (int i = 0; i < 10; ++i) {
            const int idx = i * 256 + tid;
            cw3s[idx] = cw3[(size_t)l * DD * NCLS + idx];
            rw3s[idx] = rw3[(size_t)l * DD * NCLS + idx];
        }
        __syncthreads();

        const bool active = tid < 32 * NCLS;
        int m = 0, c = 0, b = 0, q = 0;
        if (active) {
            m = tid / NCLS; c = tid % NCLS;
            const int gm = m0 + m;
            b = gm / QQ; q = gm % QQ;
            float dc = cb3[l * NCLS + c];
            float dr = rb3[l * NCLS + c];
            const _Float16* xr = X + m * 260;
            const _Float16* yr = Y + m * 260;
            #pragma unroll 8
            for (int k = 0; k < DD; ++k) {
                dc += (float)xr[k] * cw3s[k * NCLS + c];
                dr += (float)yr[k] * rw3s[k * NCLS + c];
            }
            st[m * NCLS + c] = dr;                 // stage reg tmp for postprocess
            out[((((size_t)0 * LVL + l) * BB + b) * QQ + q) * NCLS + c] = dc;
        }
        __syncthreads();
        if (active) {
            float coord;
            if (c == 0 || c == 1 || c == 4) {
                const int rc = (c == 4) ? 2 : c;
                float refv;
                if (l == 0)
                    refv = init_ref[((size_t)b * QQ + q) * 3 + rc];
                else
                    refv = inter_ref[((((size_t)(l - 1)) * BB + b) * QQ + q) * 3 + rc];
                const float t  = st[m * NCLS + c] + inv_sigmoid(refv);
                const float sg = 1.f / (1.f + expf(-t));
                coord = (c == 4) ? (sg * 8.f - 5.f) : (sg * 102.4f - 51.2f);
            } else {
                coord = st[m * NCLS + c];
            }
            out[((((size_t)1 * LVL + l) * BB + b) * QQ + q) * NCLS + c] = coord;
        }
    }
}

extern "C" void kernel_launch(void* const* d_in, const int* in_sizes, int n_in,
                              void* d_out, int out_size, void* d_ws, size_t ws_size,
                              hipStream_t stream) {
    const float* hs        = (const float*)d_in[0];
    const float* init_ref  = (const float*)d_in[1];
    const float* inter_ref = (const float*)d_in[2];
    const float* cw1 = (const float*)d_in[3];
    const float* cb1 = (const float*)d_in[4];
    const float* g1  = (const float*)d_in[5];
    const float* b1  = (const float*)d_in[6];
    const float* cw2 = (const float*)d_in[7];
    const float* cb2 = (const float*)d_in[8];
    const float* g2  = (const float*)d_in[9];
    const float* b2  = (const float*)d_in[10];
    const float* cw3 = (const float*)d_in[11];
    const float* cb3 = (const float*)d_in[12];
    const float* rw1 = (const float*)d_in[13];
    const float* rb1 = (const float*)d_in[14];
    const float* rw2 = (const float*)d_in[15];
    const float* rb2 = (const float*)d_in[16];
    const float* rw3 = (const float*)d_in[17];
    const float* rb3 = (const float*)d_in[18];
    float* out = (float*)d_out;

    const int blocks = LVL * (QQ * BB / 32);   // 6 * 900 = 5400
    bevhead_kernel<<<dim3(blocks), dim3(256), 0, stream>>>(
        hs, init_ref, inter_ref,
        cw1, cb1, g1, b1, cw2, cb2, g2, b2, cw3, cb3,
        rw1, rb1, rw2, rb2, rw3, rb3, out);
}
